// MultiHeadAttention_50749333570111
// MI455X (gfx1250) — compile-verified, matrix operand order checked by reading, not run on hardware
//
#include <hip/hip_runtime.h>

typedef __bf16 bf16;
typedef __attribute__((ext_vector_type(16))) __bf16 v16bf;
typedef __attribute__((ext_vector_type(8)))  __bf16 v8bf;
typedef __attribute__((ext_vector_type(8)))  float  v8f;
typedef __attribute__((ext_vector_type(4)))  float  v4f;

#define S_LEN  2048
#define BATCH  2
#define DMODEL 1024
#define NHEADS 16
#define HDIM   64

// 0.125 * log2(e): softmax scale folded into q so attention can use exp2
#define QSCALE 0.18033688011112042f
#define LOG2_10000 13.287712379549449f

// LDS row padding (keeps 16B alignment, destaggers banks: 72*2=144B, 40*2=80B)
#define KROW 72
#define VROW 40

static __device__ __forceinline__ v8f wmma_bf16(v16bf a, v16bf b, v8f c) {
  return __builtin_amdgcn_wmma_f32_16x16x32_bf16(false, a, false, b, (short)0, c,
                                                 false, false);
}

static __device__ __forceinline__ v16bf cat8(v8bf lo, v8bf hi) {
  v16bf r;
#pragma unroll
  for (int i = 0; i < 8; ++i) { r[i] = lo[i]; r[i + 8] = hi[i]; }
  return r;
}

static __device__ __forceinline__ v8f zero8() {
  v8f r;
#pragma unroll
  for (int i = 0; i < 8; ++i) r[i] = 0.0f;
  return r;
}

static __device__ __forceinline__ void scale8(v8f& v, float a) {
#pragma unroll
  for (int i = 0; i < 8; ++i) v[i] *= a;
}

// --- gfx1250 async DMA: global -> LDS, 16B per lane, tracked by ASYNCcnt ----
static __device__ __forceinline__ void async_copy_b128(const void* gsrc, void* lds_dst) {
  unsigned loff = (unsigned)(uintptr_t)lds_dst;  // low 32 bits = LDS byte address
  asm volatile("global_load_async_to_lds_b128 %0, %1, off"
               :: "v"(loff), "v"(gsrc) : "memory");
}

static __device__ __forceinline__ void wait_async0() {
#if __has_builtin(__builtin_amdgcn_s_wait_asynccnt)
  __builtin_amdgcn_s_wait_asynccnt(0);
#else
  asm volatile("s_wait_asynccnt 0x0" ::: "memory");
#endif
}

// ---------------------------------------------------------------------------
// fp32 -> bf16 weight conversion
// ---------------------------------------------------------------------------
__global__ __launch_bounds__(256) void cvt_f32_bf16(const float* __restrict__ src,
                                                    bf16* __restrict__ dst, int n) {
  int i = (blockIdx.x * blockDim.x + threadIdx.x) * 8;
  if (i + 8 <= n) {
    v4f a = *(const v4f*)(src + i);
    v4f b = *(const v4f*)(src + i + 4);
    v8bf o;
#pragma unroll
    for (int j = 0; j < 4; ++j) { o[j] = (bf16)a[j]; o[j + 4] = (bf16)b[j]; }
    *(v8bf*)(dst + i) = o;
  }
}

// ---------------------------------------------------------------------------
// QKV projection: out = X @ W^T + b, 16x64 tile per wave (one full head wide,
// 4 WMMA accumulators share each A fetch). mode: 0=q (rope+scale), 1=k (rope),
// 2=v (transposed store to vT[b,h,d,s]).
// ---------------------------------------------------------------------------
__global__ __launch_bounds__(256) void proj_kernel(
    const float* __restrict__ X, const bf16* __restrict__ W,
    const float* __restrict__ bias, bf16* __restrict__ out_std,
    bf16* __restrict__ out_tr, int mode) {
  const int lane = threadIdx.x;
  const int hh = lane >> 4;
  const int ln = lane & 15;
  const int w = blockIdx.x * blockDim.y + threadIdx.y;  // 0..4095
  const int b  = w >> 11;
  const int sT = (w >> 4) & 127;
  const int nG = w & 15;            // head index (64 cols == one head)
  const int s0 = sT * 16, n0 = nG * 64;

  v8f acc[4];
#pragma unroll
  for (int j = 0; j < 4; ++j) {
    float bv = bias[n0 + j * 16 + ln];
#pragma unroll
    for (int i = 0; i < 8; ++i) acc[j][i] = bv;
  }

  const float* Arow = X + ((size_t)(s0 + ln) * BATCH + b) * DMODEL;
  const bf16* Br[4];
#pragma unroll
  for (int j = 0; j < 4; ++j) Br[j] = W + (size_t)(n0 + j * 16 + ln) * DMODEL;

  for (int kk = 0; kk < DMODEL; kk += 32) {
    const int kb = kk + hh * 8;
    v4f a0 = *(const v4f*)(Arow + kb);
    v4f a1 = *(const v4f*)(Arow + kb + 4);
    v4f a2 = *(const v4f*)(Arow + kb + 16);
    v4f a3 = *(const v4f*)(Arow + kb + 20);
    v16bf A;
#pragma unroll
    for (int i = 0; i < 4; ++i) {
      A[i] = (bf16)a0[i]; A[i + 4] = (bf16)a1[i];
      A[i + 8] = (bf16)a2[i]; A[i + 12] = (bf16)a3[i];
    }
#pragma unroll
    for (int j = 0; j < 4; ++j) {
      v16bf Bm = cat8(*(const v8bf*)(Br[j] + kb), *(const v8bf*)(Br[j] + kb + 16));
      acc[j] = wmma_bf16(A, Bm, acc[j]);
    }
  }

  const size_t bh = (size_t)b * NHEADS + nG;
  if (mode == 2) {
#pragma unroll
    for (int j = 0; j < 4; ++j) {
      const int d = j * 16 + ln;
      v8bf st;
#pragma unroll
      for (int r = 0; r < 8; ++r) st[r] = (bf16)acc[j][r];
      *(v8bf*)(out_tr + (bh * HDIM + d) * S_LEN + s0 + hh * 8) = st;
    }
  } else {
    const float scale = (mode == 0) ? QSCALE : 1.0f;
    bf16* dst = out_std + bh * ((size_t)S_LEN * HDIM);
#pragma unroll
    for (int j = 0; j < 4; ++j) {
      const int d = j * 16 + ln;
      if (j < 2) {  // rotary half (d < 32); uniform branch across the wave
        const float invf = exp2f(-(float)(d & ~1) * (LOG2_10000 / 32.0f));
        const float sgn = (d & 1) ? 1.0f : -1.0f;
#pragma unroll
        for (int r = 0; r < 8; ++r) {
          float x = acc[j][r];
          float xn = __shfl_xor(x, 1, 32);
          float ang = (float)(s0 + r + hh * 8) * invf;
          float sn, cs;
          __sincosf(ang, &sn, &cs);
          acc[j][r] = x * cs + sgn * xn * sn;
        }
      }
#pragma unroll
      for (int r = 0; r < 8; ++r)
        dst[(size_t)(s0 + r + hh * 8) * HDIM + d] = (bf16)(acc[j][r] * scale);
    }
  }
}

// ---------------------------------------------------------------------------
// Flash attention. Block = 8 waves = 128 q rows of one (b,h). K/V 32-k tiles
// are DMA'd to LDS with global_load_async_to_lds_b128 (double buffered) and
// shared by all 8 waves. Scores computed TRANSPOSED so each lane owns one q
// index: softmax stats, prob repack, and O^T rescale are all lane-local.
// ---------------------------------------------------------------------------
__global__ __launch_bounds__(256) void attn_kernel(
    const bf16* __restrict__ q_ws, const bf16* __restrict__ k_ws,
    const bf16* __restrict__ vT_ws, bf16* __restrict__ ctx) {
  __shared__ bf16 Kb[2][32 * KROW];  // [k_local][d], row padded to 72
  __shared__ bf16 Vb[2][64 * VROW];  // [d][k_local], row padded to 40

  const int lane = threadIdx.x;
  const int wv = threadIdx.y;
  const int hh = lane >> 4;
  const int ln = lane & 15;
  const int bh = blockIdx.x >> 4;   // 0..31
  const int qblk = blockIdx.x & 15;
  const int b = bh >> 4;
  const int h = bh & 15;
  const int q0 = qblk * 128 + wv * 16;
  const int tid = wv * 32 + lane;   // 0..255

  const bf16* qbase = q_ws + (size_t)bh * S_LEN * HDIM;
  const char* kbytes = (const char*)(k_ws + (size_t)bh * S_LEN * HDIM);
  const char* vbytes = (const char*)(vT_ws + (size_t)bh * HDIM * S_LEN);

  // DMA chunk assignment (16B per async op, 2 ops/thread/step = 8KB/block)
  const int krow_g = tid >> 3, kcol_b = (tid & 7) * 16;    // K: 32 rows x 128B
  const int vrow_g = tid >> 2, vcol_b = (tid & 3) * 16;    // V: 64 rows x 64B

  // q tile as B operand (lane = q index), hoisted out of the k loop
  const bf16* qrow = qbase + (size_t)(q0 + ln) * HDIM + hh * 8;
  const v16bf Bq0 = cat8(*(const v8bf*)(qrow),      *(const v8bf*)(qrow + 16));
  const v16bf Bq1 = cat8(*(const v8bf*)(qrow + 32), *(const v8bf*)(qrow + 48));

  v8f o0 = zero8(), o1 = zero8(), o2 = zero8(), o3 = zero8();
  float m_run = -1e30f, l_run = 0.0f;

  // prefetch first K/V tile into buffer 0
  async_copy_b128(kbytes + (size_t)krow_g * 128 + kcol_b,
                  (char*)&Kb[0][0] + krow_g * (KROW * 2) + kcol_b);
  async_copy_b128(vbytes + (size_t)vrow_g * (S_LEN * 2) + vcol_b,
                  (char*)&Vb[0][0] + vrow_g * (VROW * 2) + vcol_b);

  for (int kt = 0; kt < S_LEN; kt += 32) {
    const int bb = (kt >> 5) & 1;
    wait_async0();        // my DMA into buffer bb has landed
    __syncthreads();      // everyone's DMA landed; everyone done reading bb^1
    if (kt + 32 < S_LEN) {
      const int nb = bb ^ 1;
      async_copy_b128(kbytes + (size_t)(kt + 32 + krow_g) * 128 + kcol_b,
                      (char*)&Kb[nb][0] + krow_g * (KROW * 2) + kcol_b);
      async_copy_b128(vbytes + (size_t)vrow_g * (S_LEN * 2) + (kt + 32) * 2 + vcol_b,
                      (char*)&Vb[nb][0] + vrow_g * (VROW * 2) + vcol_b);
    }

    // S^T subtiles from LDS: rows k=kt..kt+15 (c0), kt+16..kt+31 (c1)
    const bf16* kr0 = &Kb[bb][(0 + ln) * KROW + hh * 8];
    const bf16* kr1 = &Kb[bb][(16 + ln) * KROW + hh * 8];
    v8f c0 = zero8(), c1 = zero8();
    c0 = wmma_bf16(cat8(*(const v8bf*)(kr0),      *(const v8bf*)(kr0 + 16)), Bq0, c0);
    c0 = wmma_bf16(cat8(*(const v8bf*)(kr0 + 32), *(const v8bf*)(kr0 + 48)), Bq1, c0);
    c1 = wmma_bf16(cat8(*(const v8bf*)(kr1),      *(const v8bf*)(kr1 + 16)), Bq0, c1);
    c1 = wmma_bf16(cat8(*(const v8bf*)(kr1 + 32), *(const v8bf*)(kr1 + 48)), Bq1, c1);

    // online softmax (base-2 domain; scale folded into q)
    float mx = c0[0];
#pragma unroll
    for (int r = 0; r < 8; ++r) { mx = fmaxf(mx, c0[r]); mx = fmaxf(mx, c1[r]); }
    mx = fmaxf(mx, __shfl_xor(mx, 16, 32));
    const float m_new = fmaxf(m_run, mx);
    const float alpha = exp2f(m_run - m_new);

    float ls = 0.0f;
    v16bf P;
#pragma unroll
    for (int r = 0; r < 8; ++r) {
      float p0 = exp2f(c0[r] - m_new);
      float p1 = exp2f(c1[r] - m_new);
      ls += p0 + p1;
      P[r] = (bf16)p0;
      P[r + 8] = (bf16)p1;
    }
    ls += __shfl_xor(ls, 16, 32);
    l_run = l_run * alpha + ls;
    m_run = m_new;

    scale8(o0, alpha); scale8(o1, alpha); scale8(o2, alpha); scale8(o3, alpha);
    {
      const bf16* vr0 = &Vb[bb][(0 * 16 + ln) * VROW + hh * 8];
      const bf16* vr1 = &Vb[bb][(1 * 16 + ln) * VROW + hh * 8];
      const bf16* vr2 = &Vb[bb][(2 * 16 + ln) * VROW + hh * 8];
      const bf16* vr3 = &Vb[bb][(3 * 16 + ln) * VROW + hh * 8];
      o0 = wmma_bf16(cat8(*(const v8bf*)(vr0), *(const v8bf*)(vr0 + 16)), P, o0);
      o1 = wmma_bf16(cat8(*(const v8bf*)(vr1), *(const v8bf*)(vr1 + 16)), P, o1);
      o2 = wmma_bf16(cat8(*(const v8bf*)(vr2), *(const v8bf*)(vr2 + 16)), P, o2);
      o3 = wmma_bf16(cat8(*(const v8bf*)(vr3), *(const v8bf*)(vr3 + 16)), P, o3);
    }
  }

  const float inv_l = 1.0f / l_run;
  bf16* cp = ctx + (((size_t)b * S_LEN + q0 + ln) * NHEADS + h) * HDIM + hh * 8;
  v8bf s0v, s1v, s2v, s3v;
#pragma unroll
  for (int r = 0; r < 8; ++r) {
    s0v[r] = (bf16)(o0[r] * inv_l);
    s1v[r] = (bf16)(o1[r] * inv_l);
    s2v[r] = (bf16)(o2[r] * inv_l);
    s3v[r] = (bf16)(o3[r] * inv_l);
  }
  *(v8bf*)(cp + 0)  = s0v;
  *(v8bf*)(cp + 16) = s1v;
  *(v8bf*)(cp + 32) = s2v;
  *(v8bf*)(cp + 48) = s3v;
}

// ---------------------------------------------------------------------------
// Output projection: out(S,B,D) = ctx(b,s,:) @ Wo^T + bo, 16x64 per wave
// ---------------------------------------------------------------------------
__global__ __launch_bounds__(256) void oproj_kernel(
    const bf16* __restrict__ ctx, const bf16* __restrict__ Wo,
    const float* __restrict__ bo, float* __restrict__ out) {
  const int lane = threadIdx.x;
  const int hh = lane >> 4;
  const int ln = lane & 15;
  const int w = blockIdx.x * blockDim.y + threadIdx.y;  // 0..4095
  const int b  = w >> 11;
  const int sT = (w >> 4) & 127;
  const int nG = w & 15;
  const int s0 = sT * 16, n0 = nG * 64;

  v8f acc[4];
#pragma unroll
  for (int j = 0; j < 4; ++j) {
    float bv = bo[n0 + j * 16 + ln];
#pragma unroll
    for (int i = 0; i < 8; ++i) acc[j][i] = bv;
  }

  const bf16* Arow = ctx + ((size_t)b * S_LEN + s0 + ln) * DMODEL;
  const bf16* Br[4];
#pragma unroll
  for (int j = 0; j < 4; ++j) Br[j] = Wo + (size_t)(n0 + j * 16 + ln) * DMODEL;

  for (int kk = 0; kk < DMODEL; kk += 32) {
    const int kb = kk + hh * 8;
    v16bf A = cat8(*(const v8bf*)(Arow + kb), *(const v8bf*)(Arow + kb + 16));
#pragma unroll
    for (int j = 0; j < 4; ++j) {
      v16bf Bm = cat8(*(const v8bf*)(Br[j] + kb), *(const v8bf*)(Br[j] + kb + 16));
      acc[j] = wmma_bf16(A, Bm, acc[j]);
    }
  }

#pragma unroll
  for (int j = 0; j < 4; ++j)
#pragma unroll
    for (int r = 0; r < 8; ++r)
      out[((size_t)(s0 + r + hh * 8) * BATCH + b) * DMODEL + n0 + j * 16 + ln] =
          acc[j][r];
}

// ---------------------------------------------------------------------------
extern "C" void kernel_launch(void* const* d_in, const int* in_sizes, int n_in,
                              void* d_out, int out_size, void* d_ws, size_t ws_size,
                              hipStream_t stream) {
  (void)in_sizes; (void)n_in; (void)out_size; (void)ws_size;
  const float* Q  = (const float*)d_in[0];
  const float* K  = (const float*)d_in[1];
  const float* V  = (const float*)d_in[2];
  const float* Wq = (const float*)d_in[3];
  const float* bq = (const float*)d_in[4];
  const float* Wk = (const float*)d_in[5];
  const float* bk = (const float*)d_in[6];
  const float* Wv = (const float*)d_in[7];
  const float* bv = (const float*)d_in[8];
  const float* Wo = (const float*)d_in[9];
  const float* bo = (const float*)d_in[10];
  float* out = (float*)d_out;

  const size_t WELEM = (size_t)DMODEL * DMODEL;                 // 1,048,576
  const size_t QKV   = (size_t)BATCH * NHEADS * S_LEN * HDIM;   // 4,194,304
  bf16* ws    = (bf16*)d_ws;
  bf16* Wq_bf = ws;
  bf16* Wk_bf = Wq_bf + WELEM;
  bf16* Wv_bf = Wk_bf + WELEM;
  bf16* Wo_bf = Wv_bf + WELEM;
  bf16* q_ws  = Wo_bf + WELEM;
  bf16* k_ws  = q_ws + QKV;
  bf16* vT_ws = k_ws + QKV;
  bf16* ctx   = vT_ws + QKV;   // total ws use: 40 MB

  cvt_f32_bf16<<<512, 256, 0, stream>>>(Wq, Wq_bf, (int)WELEM);
  cvt_f32_bf16<<<512, 256, 0, stream>>>(Wk, Wk_bf, (int)WELEM);
  cvt_f32_bf16<<<512, 256, 0, stream>>>(Wv, Wv_bf, (int)WELEM);
  cvt_f32_bf16<<<512, 256, 0, stream>>>(Wo, Wo_bf, (int)WELEM);

  dim3 blk(32, 8);  // 8 waves of 32
  proj_kernel<<<512, blk, 0, stream>>>(Q, Wq_bf, bq, q_ws, nullptr, 0);
  proj_kernel<<<512, blk, 0, stream>>>(K, Wk_bf, bk, k_ws, nullptr, 1);
  proj_kernel<<<512, blk, 0, stream>>>(V, Wv_bf, bv, nullptr, vT_ws, 2);
  attn_kernel<<<512, blk, 0, stream>>>(q_ws, k_ws, vT_ws, ctx);
  oproj_kernel<<<512, blk, 0, stream>>>(ctx, Wo_bf, bo, out);
}
